// CausalSelfAttention_17428977287931
// MI455X (gfx1250) — matrix operand order repaired and checked by reading, not compiled, not run on hardware
//
#include <hip/hip_runtime.h>
#include <cstdint>

#define DEVINL __device__ __forceinline__

typedef __bf16 bf16;
typedef __attribute__((ext_vector_type(16))) __bf16 v16bf;
typedef __attribute__((ext_vector_type(8)))  __bf16 v8bf;
typedef __attribute__((ext_vector_type(8)))  float  v8f;
typedef __attribute__((ext_vector_type(4)))  unsigned int v4u;
typedef __attribute__((ext_vector_type(8)))  int v8i;
typedef __attribute__((ext_vector_type(4)))  int v4i;

constexpr int Bc = 4, Sc = 2048, DIMc = 1024, Hc = 16, KVHc = 4, HDc = 64;
constexpr int KVDIMc = KVHc * HDc;  // 256
constexpr float RMS_EPS_F = 1.1920928955078125e-07f;

// ---------------------------------------------------------------------------
// WMMA helpers (V_WMMA_F32_16X16X32_BF16, wave32)
// ---------------------------------------------------------------------------
DEVINL v8f wmma_bf16(v16bf a, v16bf b, v8f c) {
  // 8 args: (neg_a, A, neg_b, B, c_mod, C, reuse_a, reuse_b)
  return __builtin_amdgcn_wmma_f32_16x16x32_bf16(false, a, false, b, (short)0, c,
                                                 false, false);
}

// A-fragment (16x32, bf16): lane half hl: elems 0..15 -> K = k0 + hl*16 + j
// (contiguous).  `row` points at A row (lane&15).
DEVINL v16bf load_a_frag(const bf16* row, int k0, int lane) {
  const int hl = (lane >> 4) & 1;
  v8bf lo = *(const v8bf*)(row + k0 + hl * 16);
  v8bf hi = *(const v8bf*)(row + k0 + hl * 16 + 8);
  v16bf a;
#pragma unroll
  for (int j = 0; j < 8; ++j) { a[j] = lo[j]; a[j + 8] = hi[j]; }
  return a;
}

// ---------------------------------------------------------------------------
// Tensor Data Mover: async 2D tile load Global -> LDS with LDS padding.
// D# layout per CDNA5 ISA (group0 128b, group1 256b;
// groups 2/3 zero for a 2D tensor). data_size = 2 bytes (bf16).
// pad: after (2 << pad_ic) DWORDs stored, skip (pad_ac + 1) DWORDs in LDS.
// ---------------------------------------------------------------------------
DEVINL void tdm_load_2d_bf16(uint32_t lds_byte_addr, const void* gaddr,
                             uint32_t tile_w, uint32_t tile_h,
                             uint32_t tensor_w, uint32_t tensor_h,
                             uint32_t row_stride_elems,
                             uint32_t pad_ic, uint32_t pad_ac, bool pad_en) {
  uint64_t ga = (uint64_t)(uintptr_t)gaddr;
  v4u g0;
  g0[0] = 1u;                                          // count=1, user mode
  g0[1] = lds_byte_addr;                               // lds_addr [63:32]
  g0[2] = (uint32_t)(ga & 0xFFFFFFFFu);                // global_addr[31:0]
  g0[3] = ((uint32_t)(ga >> 32) & 0x01FFFFFFu) | (2u << 30);  // addr[56:32]|type=2

  uint32_t w0 = (1u << 16);                            // data_size = 1 -> 2 bytes
  if (pad_en) w0 |= (1u << 20) | (pad_ic << 22) | (pad_ac << 25);
  v8i g1;
  g1[0] = (int)w0;
  g1[1] = (int)((tensor_w & 0xFFFFu) << 16);                    // dim0[15:0] @63:48
  g1[2] = (int)(((tensor_w >> 16) & 0xFFFFu) |
                ((tensor_h & 0xFFFFu) << 16));                  // dim0 hi | dim1 lo
  g1[3] = (int)(((tensor_h >> 16) & 0xFFFFu) |
                ((tile_w & 0xFFFFu) << 16));                    // dim1 hi | tile0
  g1[4] = (int)(tile_h & 0xFFFFu);                              // tile1 | tile2=0
  g1[5] = (int)row_stride_elems;                                // dim0_stride[31:0]
  g1[6] = 0;                                                    // stride hi, dim1_stride lo
  g1[7] = 0;
  v4i g2 = {0, 0, 0, 0};
  v4i g3 = {0, 0, 0, 0};
#if defined(__clang_major__) && (__clang_major__ >= 23)
  v8i gx = {0, 0, 0, 0, 0, 0, 0, 0};
  __builtin_amdgcn_tensor_load_to_lds(g0, g1, g2, g3, gx, 0);
#else
  __builtin_amdgcn_tensor_load_to_lds(g0, g1, g2, g3, 0);
#endif
}

// ---------------------------------------------------------------------------
// fp32 -> bf16 cast
// ---------------------------------------------------------------------------
__global__ __launch_bounds__(256) void cast_f32_to_bf16(const float* __restrict__ s,
                                                        bf16* __restrict__ d, int n) {
  int i = blockIdx.x * blockDim.x + threadIdx.x;
  if (i < n) d[i] = (bf16)s[i];
}

// ---------------------------------------------------------------------------
// C[m][n] = sum_k A[m][k] * W[n][k]   (A: MxK bf16, W: NxK bf16)
// Block = 256 threads = 8 waves; wave computes a 16(M) x 64(N) tile.
// B-fragment: b[i] = W[nbase+lane&15][k0 + hl*16 + i]  (16 contiguous bf16)
// ---------------------------------------------------------------------------
template <bool F32OUT>
__global__ __launch_bounds__(256)
void gemm_xwt(const bf16* __restrict__ A, const bf16* __restrict__ W,
              void* __restrict__ Cout, int M, int N, int K) {
  const int lane = threadIdx.x & 31;
  const int wave = threadIdx.x >> 5;
  const int hl = (lane >> 4) & 1;
  const int ln = lane & 15;
  const int mtile = blockIdx.x * 128 + wave * 16;
  const int nbase = blockIdx.y * 64;
  const bf16* arow = A + (size_t)(mtile + ln) * K;

  v8f acc[4] = {};
  for (int k0 = 0; k0 < K; k0 += 32) {
    __builtin_prefetch(arow + k0 + 128, 0, 1);
    v16bf a = load_a_frag(arow, k0, lane);
#pragma unroll
    for (int t = 0; t < 4; ++t) {
      const bf16* bp = W + (size_t)(nbase + t * 16 + ln) * K + k0 + hl * 16;
      v16bf b = *(const v16bf*)bp;
      acc[t] = wmma_bf16(a, b, acc[t]);
    }
  }
  // C/D layout: VGPR r holds row (8*hl + r), column = lane&15 within each tile.
#pragma unroll
  for (int t = 0; t < 4; ++t) {
#pragma unroll
    for (int r = 0; r < 8; ++r) {
      size_t row = (size_t)(mtile + hl * 8 + r);
      float v = acc[t][r];
      if constexpr (F32OUT)
        ((float*)Cout)[row * N + nbase + t * 16 + ln] = v;
      else
        ((bf16*)Cout)[row * N + nbase + t * 16 + ln] = (bf16)v;
    }
  }
}

// ---------------------------------------------------------------------------
// Per-(token, head) RMS-norm + RoPE (+ gain * 1/sqrt(HD) for q). One wave/row.
// Lane j handles elements j and j+32 of the 64-wide head vector.
// ---------------------------------------------------------------------------
__global__ __launch_bounds__(256)
void qk_norm_rope(bf16* __restrict__ buf, int nH, int totalRows,
                  const float* __restrict__ gain, float extraScale) {
  int wid = blockIdx.x * 8 + (threadIdx.x >> 5);
  int lane = threadIdx.x & 31;
  if (wid >= totalRows) return;
  int head = wid % nH;
  int tok = wid / nH;
  int pos = tok % Sc;
  bf16* row = buf + (size_t)tok * (nH * HDc) + head * HDc;

  float x1 = (float)row[lane];
  float x2 = (float)row[lane + 32];
  float ss = x1 * x1 + x2 * x2;
#pragma unroll
  for (int m = 16; m >= 1; m >>= 1) ss += __shfl_xor(ss, m, 32);
  float rn = rsqrtf(ss * (1.0f / 64.0f) + RMS_EPS_F);
  x1 *= rn; x2 *= rn;

  // inv_freq[j] = 10000^(-j/32); ln(10000) = 9.210340371976184
  float inv_freq = __expf(-(float)lane * (9.210340371976184f / 32.0f));
  float ang = (float)pos * inv_freq;
  float sv, cv;
  sincosf(ang, &sv, &cv);
  float g = (gain != nullptr) ? gain[head] * extraScale : 1.0f;
  row[lane]      = (bf16)((x1 * cv + x2 * sv) * g);
  row[lane + 32] = (bf16)((-x1 * sv + x2 * cv) * g);
}

// ---------------------------------------------------------------------------
// Flash attention. Block = (128 q-rows, head h, batch b), 8 waves, each wave
// owns 16 q rows. Key tile (64x64 bf16) staged to LDS by the Tensor Data
// Mover (wave 0 issues, TENSORcnt waited, padded to the 80-elem pitch the
// WMMA B-fragments read). V tile staged transposed by all threads. Online
// softmax in f32, P->LDS->A-fragment, O += P @ V via WMMA.
// q is pre-scaled by gain/sqrt(HD).
// ---------------------------------------------------------------------------
__global__ __launch_bounds__(256)
void attn_kernel(const bf16* __restrict__ qb, const bf16* __restrict__ kb,
                 const bf16* __restrict__ vb, bf16* __restrict__ yb) {
  const int lane = threadIdx.x & 31;
  const int wave = threadIdx.x >> 5;
  const int hl = (lane >> 4) & 1;
  const int ln = lane & 15;
  const int qblk = blockIdx.x;
  const int h = blockIdx.y;
  const int b = blockIdx.z;
  const int kvh = h >> 2;            // G = 4
  const int q0 = qblk * 128;
  const int myq = q0 + wave * 16;

  // pitch 80 bf16 = 160 B (multiple of 32 B -> v16bf stays aligned)
  __shared__ __align__(32) bf16 kt[64 * 80];          // [key][d]   (TDM-filled)
  __shared__ __align__(32) bf16 vt[64 * 80];          // [d][key]   (transposed)
  __shared__ __align__(32) bf16 pt[8][16 * 80];       // per-wave P tile

  // LDS byte address of kt: flat-pointer low 32 bits are the LDS address.
  const uint32_t kt_lds = (uint32_t)(uintptr_t)(&kt[0]);

  const bf16* qrow = qb + ((size_t)(b * Sc + myq + ln)) * DIMc + h * HDc;
  v16bf qa0 = load_a_frag(qrow, 0, lane);
  v16bf qa1 = load_a_frag(qrow, 32, lane);

  v8f o[4] = {};
  float mrow[8], lrow[8];
#pragma unroll
  for (int r = 0; r < 8; ++r) { mrow[r] = -1e30f; lrow[r] = 0.f; }

  const int ntiles = 2 * qblk + 2;       // ceil((q0+128)/64)
  const int skey = threadIdx.x >> 2;     // staging: key row 0..63
  const int sseg = (threadIdx.x & 3) * 16;

  for (int t = 0; t < ntiles; ++t) {
    const int kbase = t * 64;
    __syncthreads();
    if (wave == 0) {
      // K tile: 64 rows x 64 bf16, global row stride 256 elems ->
      // LDS rows of 32 DWORDs + 8 DWORDs pad (pitch 80 bf16).
      const bf16* gk = kb + (size_t)(b * Sc + kbase) * KVDIMc + kvh * HDc;
      tdm_load_2d_bf16(kt_lds, gk, /*tile*/64, 64, /*tensor*/64, 64,
                       /*stride*/KVDIMc, /*pad_ic: 2<<4=32 dw*/4,
                       /*pad_ac: 7+1=8 dw*/7, true);
    }
    {
      // V tile staged transposed: vt[d][key] = V[key][d]
      const size_t srow =
          (size_t)(b * Sc + kbase + skey) * KVDIMc + kvh * HDc + sseg;
      v16bf tv = *(const v16bf*)(vb + srow);
#pragma unroll
      for (int j = 0; j < 16; ++j) vt[(sseg + j) * 80 + skey] = tv[j];
    }
    if (wave == 0) __builtin_amdgcn_s_wait_tensorcnt(0);
    __syncthreads();

    if (kbase <= myq + 15) {
      // S = Q (16 x 64) @ K^T (64 keys): B-frag b[i] = K[key=nt*16+ln][k0+hl*16+i]
      v8f sacc[4] = {};
#pragma unroll
      for (int nt = 0; nt < 4; ++nt) {
        v16bf bk0 = *(const v16bf*)&kt[(nt * 16 + ln) * 80 + hl * 16];
        sacc[nt] = wmma_bf16(qa0, bk0, sacc[nt]);
        v16bf bk1 = *(const v16bf*)&kt[(nt * 16 + ln) * 80 + 32 + hl * 16];
        sacc[nt] = wmma_bf16(qa1, bk1, sacc[nt]);
      }
      // causal mask: this lane's key col, rows 8*hl + r
#pragma unroll
      for (int nt = 0; nt < 4; ++nt) {
        const int kc = kbase + nt * 16 + ln;
#pragma unroll
        for (int r = 0; r < 8; ++r)
          if (kc > myq + hl * 8 + r) sacc[nt][r] = -1e30f;
      }
      // online softmax (row stats live per lane-half; reduce over 16 lanes)
#pragma unroll
      for (int r = 0; r < 8; ++r) {
        float tm = sacc[0][r];
#pragma unroll
        for (int nt = 1; nt < 4; ++nt) tm = fmaxf(tm, sacc[nt][r]);
#pragma unroll
        for (int m = 8; m >= 1; m >>= 1) tm = fmaxf(tm, __shfl_xor(tm, m, 16));
        float nm = fmaxf(mrow[r], tm);
        float alpha = __expf(mrow[r] - nm);
        mrow[r] = nm;
        lrow[r] *= alpha;
#pragma unroll
        for (int f = 0; f < 4; ++f) o[f][r] *= alpha;
        float rs = 0.f;
#pragma unroll
        for (int nt = 0; nt < 4; ++nt) {
          float p = __expf(sacc[nt][r] - nm);
          sacc[nt][r] = p;
          rs += p;
        }
#pragma unroll
        for (int m = 8; m >= 1; m >>= 1) rs += __shfl_xor(rs, m, 16);
        lrow[r] += rs;
#pragma unroll
        for (int nt = 0; nt < 4; ++nt)
          pt[wave][(hl * 8 + r) * 80 + nt * 16 + ln] = (bf16)sacc[nt][r];
      }
      // intra-wave LDS RAW: LDS ops are in-order, but be explicit
      asm volatile("s_wait_dscnt 0" ::: "memory");

      // O += P (16 x 64) @ V (64 x 64); V B-frag from transposed vt
      const bf16* prow = &pt[wave][ln * 80];
      v16bf pa0 = load_a_frag(prow, 0, lane);
      v16bf pa1 = load_a_frag(prow, 32, lane);
#pragma unroll
      for (int nt = 0; nt < 4; ++nt) {
        v16bf vb0 = *(const v16bf*)&vt[(nt * 16 + ln) * 80 + hl * 16];
        o[nt] = wmma_bf16(pa0, vb0, o[nt]);
        v16bf vb1 = *(const v16bf*)&vt[(nt * 16 + ln) * 80 + 32 + hl * 16];
        o[nt] = wmma_bf16(pa1, vb1, o[nt]);
      }
    }
  }
  // normalize and write y[b, s, h*64 + d]
#pragma unroll
  for (int f = 0; f < 4; ++f) {
#pragma unroll
    for (int r = 0; r < 8; ++r) {
      float val = o[f][r] / lrow[r];
      yb[((size_t)(b * Sc + myq + hl * 8 + r)) * DIMc + h * HDc + f * 16 + ln] =
          (bf16)val;
    }
  }
}

// ---------------------------------------------------------------------------
// launch
// ---------------------------------------------------------------------------
extern "C" void kernel_launch(void* const* d_in, const int* in_sizes, int n_in,
                              void* d_out, int out_size, void* d_ws, size_t ws_size,
                              hipStream_t stream) {
  (void)in_sizes; (void)n_in; (void)out_size; (void)ws_size;
  const float* x  = (const float*)d_in[0];
  const float* Wq = (const float*)d_in[1];
  const float* Wk = (const float*)d_in[2];
  const float* Wv = (const float*)d_in[3];
  const float* Wp = (const float*)d_in[4];
  const float* qg = (const float*)d_in[5];

  const int M = Bc * Sc;  // 8192 token rows

  size_t off = 0;
  auto carve = [&](size_t elems) -> bf16* {
    bf16* p = (bf16*)((char*)d_ws + off);
    off += (elems * 2 + 255) & ~(size_t)255;
    return p;
  };
  bf16* xb  = carve((size_t)M * DIMc);
  bf16* wqb = carve((size_t)DIMc * DIMc);
  bf16* wkb = carve((size_t)KVDIMc * DIMc);
  bf16* wvb = carve((size_t)KVDIMc * DIMc);
  bf16* wpb = carve((size_t)DIMc * DIMc);
  bf16* qbf = carve((size_t)M * DIMc);
  bf16* kbf = carve((size_t)M * KVDIMc);
  bf16* vbf = carve((size_t)M * KVDIMc);
  bf16* ybf = carve((size_t)M * DIMc);

  auto castL = [&](const float* s, bf16* d, int n) {
    cast_f32_to_bf16<<<(n + 255) / 256, 256, 0, stream>>>(s, d, n);
  };
  castL(x,  xb,  M * DIMc);
  castL(Wq, wqb, DIMc * DIMc);
  castL(Wk, wkb, KVDIMc * DIMc);
  castL(Wv, wvb, KVDIMc * DIMc);
  castL(Wp, wpb, DIMc * DIMc);

  dim3 blk(256);
  gemm_xwt<false><<<dim3(M / 128, DIMc / 64),   blk, 0, stream>>>(xb, wqb, qbf, M, DIMc,   DIMc);
  gemm_xwt<false><<<dim3(M / 128, KVDIMc / 64), blk, 0, stream>>>(xb, wkb, kbf, M, KVDIMc, DIMc);
  gemm_xwt<false><<<dim3(M / 128, KVDIMc / 64), blk, 0, stream>>>(xb, wvb, vbf, M, KVDIMc, DIMc);

  // q: fold gain * 1/sqrt(64); k: plain norm+rope
  qk_norm_rope<<<(M * Hc) / 8,   256, 0, stream>>>(qbf, Hc,   M * Hc,   qg, 0.125f);
  qk_norm_rope<<<(M * KVHc) / 8, 256, 0, stream>>>(kbf, KVHc, M * KVHc, nullptr, 1.0f);

  attn_kernel<<<dim3(Sc / 128, Hc, Bc), blk, 0, stream>>>(qbf, kbf, vbf, ybf);

  gemm_xwt<true><<<dim3(M / 128, DIMc / 64), blk, 0, stream>>>(ybf, wpb, d_out, M, DIMc, DIMc);
}
